// ObjectConceptMoE_53412213293900
// MI455X (gfx1250) — matrix operand order repaired and checked by reading, not compiled, not run on hardware
//
#include <hip/hip_runtime.h>
#include <cstdint>
#include <cstddef>

// ---------------- problem constants ----------------
#define TOK   2048   // NOBJ * N tokens
#define HD    1024   // hidden dim
#define EHD   2048   // expert hidden dim
#define NHEADS 32
#define NOBJS  4
#define NSEQ   512

// ---------------- types ----------------
typedef __attribute__((ext_vector_type(16))) __bf16         v16bf;
typedef __attribute__((ext_vector_type(8)))  float          v8f;
typedef __attribute__((ext_vector_type(8)))  unsigned short u16x8;

union Frag16 { u16x8 h[2]; v16bf v; };

__device__ __forceinline__ unsigned short f2bf(float f) {
  __bf16 b = (__bf16)f;                       // native RNE convert on gfx1250
  return __builtin_bit_cast(unsigned short, b);
}
__device__ __forceinline__ float bf2f(unsigned short h) {
  unsigned int u = ((unsigned int)h) << 16;
  return __builtin_bit_cast(float, u);
}
__device__ __forceinline__ float gelu_f(float x) {  // tanh approximation (jax default)
  return 0.5f * x * (1.0f + tanhf(0.7978845608028654f * (x + 0.044715f * x * x * x)));
}
__device__ __forceinline__ v8f wmma_bf16(const Frag16& a, const Frag16& b, v8f c) {
  return __builtin_amdgcn_wmma_f32_16x16x32_bf16(false, a.v, false, b.v, (short)0, c,
                                                 false, false);
}

// async 2x16B global -> LDS copy (CDNA5 TDM-lite path, ASYNCcnt-tracked)
__device__ __forceinline__ void async_cp32(const void* gptr, void* lptr) {
  const unsigned loff = (unsigned)(size_t)lptr;   // LDS aperture: low 32 bits = offset
  asm volatile(
      "global_load_async_to_lds_b128 %0, %1, off\n\t"
      "global_load_async_to_lds_b128 %0, %1, off offset:16"
      :: "v"(loff), "v"(gptr) : "memory");
}
__device__ __forceinline__ void async_wait0() {
  asm volatile("s_wait_asynccnt 0x0" ::: "memory");
}

// ---------------- epilogue flags ----------------
enum {
  F_GELU        = 1,   // r = gelu(acc + bias)
  F_SCALE_AFTER = 2,   // r *= scale[m]          (after gelu)
  F_SCALED_BIAS = 4,   // r = acc + scale[m]*bias[n]
  F_ADDSRC      = 8,   // r += addsrc[m,n]
  F_OUT_BF16    = 16   // store bf16 instead of f32
};

// ---------------- weight convert+transpose: fp32 [K][N] -> bf16 [N][K] --------
__global__ __launch_bounds__(256)
void wT_k(const float* __restrict__ src, unsigned short* __restrict__ dst,
          int K, int N)
{
  __shared__ unsigned short t[32][33];
  const size_t mtx = (size_t)blockIdx.z * (size_t)K * N;
  const int k0 = blockIdx.y * 32;
  const int n0 = blockIdx.x * 32;
  const int c  = threadIdx.x & 31;
  const int r0 = threadIdx.x >> 5;     // 0..7
#pragma unroll
  for (int i = 0; i < 4; ++i) {
    const int kr = r0 + i * 8;
    t[c][kr] = f2bf(src[mtx + (size_t)(k0 + kr) * N + n0 + c]);
  }
  __syncthreads();
#pragma unroll
  for (int i = 0; i < 4; ++i) {
    const int nr = r0 + i * 8;
    dst[mtx + (size_t)(n0 + nr) * K + k0 + c] = t[nr][c];
  }
}

// ---------------- bf16-WMMA GEMM: C = epi(A @ Bt^T) ----------------
// A: M x K bf16 row-major. Bt: N x K bf16 row-major (pre-transposed weights).
// M multiple of 128, N multiple of 128, K multiple of 32.
#define BM 128
#define BN 128
#define BK 32

template<int FLAGS>
__global__ __launch_bounds__(256)
void gemm_k(const unsigned short* __restrict__ A,
            const unsigned short* __restrict__ Bt,
            const float* __restrict__ bias,
            const float* __restrict__ scale, int sstr,
            const float* __restrict__ addsrc,
            float* __restrict__ Cf, unsigned short* __restrict__ Cb,
            int M, int N, int K)
{
  __shared__ __align__(16) unsigned short As[BM][BK + 8];
  __shared__ __align__(16) unsigned short Bs[BN][BK + 8];

  const int tid  = threadIdx.x;
  const int wave = tid >> 5;
  const int lane = tid & 31;
  const int r  = lane & 15;
  const int hi = lane >> 4;                  // 0 or 1
  const int mbase = blockIdx.y * BM;
  const int nbase = blockIdx.x * BN;
  const int mts = (wave & 3) * 2;            // 2 M subtiles per wave
  const int nb  = (wave >> 2) * 4;           // 4 N subtiles per wave

  v8f c[2][4] = {};

  const int srow = tid >> 1;                 // 0..127
  const int scol = (tid & 1) * 16;           // 0 or 16

  for (int k0 = 0; k0 < K; k0 += BK) {
    // stage tiles via async global->LDS (no VGPR round-trip, ASYNCcnt-tracked)
    async_cp32(A  + (size_t)(mbase + srow) * K + k0 + scol, &As[srow][scol]);
    async_cp32(Bt + (size_t)(nbase + srow) * K + k0 + scol, &Bs[srow][scol]);
    async_wait0();
    __syncthreads();

    // fragments (ISA 7.12.2 layouts)
    Frag16 a[2], b[4];
    const int akb = hi * 8;
#pragma unroll
    for (int i = 0; i < 2; ++i) {
      a[i].h[0] = *(const u16x8*)&As[(mts + i) * 16 + r][akb];
      a[i].h[1] = *(const u16x8*)&As[(mts + i) * 16 + r][akb + 16];
    }
    const int bkb = hi * 8;
#pragma unroll
    for (int j = 0; j < 4; ++j) {
      b[j].h[0] = *(const u16x8*)&Bs[(nb + j) * 16 + r][bkb];
      b[j].h[1] = *(const u16x8*)&Bs[(nb + j) * 16 + r][bkb + 16];
    }
#pragma unroll
    for (int i = 0; i < 2; ++i)
#pragma unroll
      for (int j = 0; j < 4; ++j)
        c[i][j] = wmma_bf16(a[i], b[j], c[i][j]);
    __syncthreads();
  }

  // epilogue
#pragma unroll
  for (int i = 0; i < 2; ++i) {
#pragma unroll
    for (int j = 0; j < 4; ++j) {
      const int n = nbase + (nb + j) * 16 + r;
#pragma unroll
      for (int rr = 0; rr < 8; ++rr) {
        const int m = mbase + (mts + i) * 16 + rr + hi * 8;
        float v = c[i][j][rr];
        float sc = 1.0f;
        if (FLAGS & (F_SCALE_AFTER | F_SCALED_BIAS)) sc = scale[(size_t)m * sstr];
        if (FLAGS & F_SCALED_BIAS) v += sc * bias[n];
        else                       v += bias[n];
        if (FLAGS & F_GELU)        v = gelu_f(v);
        if (FLAGS & F_SCALE_AFTER) v *= sc;
        if (FLAGS & F_ADDSRC)      v += addsrc[(size_t)m * N + n];
        if (FLAGS & F_OUT_BF16) Cb[(size_t)m * N + n] = f2bf(v);
        else                    Cf[(size_t)m * N + n] = v;
      }
    }
  }
}

// ---------------- LayerNorm (per token row), fp32 + bf16 outputs ----------------
__global__ __launch_bounds__(256)
void ln_k(const float* __restrict__ X, const float* __restrict__ g,
          const float* __restrict__ bta, float* __restrict__ Yf,
          unsigned short* __restrict__ Yb)
{
  const int row = blockIdx.x;
  const int tid = threadIdx.x;
  __shared__ float rbuf[256];
  const float* xp = X + (size_t)row * HD;
  float vals[4];
  float s = 0.f;
#pragma unroll
  for (int i = 0; i < 4; ++i) { vals[i] = xp[tid + i * 256]; s += vals[i]; }
  rbuf[tid] = s; __syncthreads();
  for (int off = 128; off > 0; off >>= 1) {
    if (tid < off) rbuf[tid] += rbuf[tid + off];
    __syncthreads();
  }
  const float mean = rbuf[0] * (1.0f / HD);
  __syncthreads();
  float vs = 0.f;
#pragma unroll
  for (int i = 0; i < 4; ++i) { float d = vals[i] - mean; vs += d * d; }
  rbuf[tid] = vs; __syncthreads();
  for (int off = 128; off > 0; off >>= 1) {
    if (tid < off) rbuf[tid] += rbuf[tid + off];
    __syncthreads();
  }
  const float inv = rsqrtf(rbuf[0] * (1.0f / HD) + 1e-5f);
#pragma unroll
  for (int i = 0; i < 4; ++i) {
    const int c = tid + i * 256;
    const float y = (vals[i] - mean) * inv * g[c] + bta[c];
    if (Yf) Yf[(size_t)row * HD + c] = y;
    Yb[(size_t)row * HD + c] = f2bf(y);
  }
}

// ---------------- attention: per (obj, head, 64-query tile) ----------------
#define QT 64
__global__ __launch_bounds__(256)
void attn_k(const unsigned short* __restrict__ Q, const unsigned short* __restrict__ Kb,
            const unsigned short* __restrict__ Vb, unsigned short* __restrict__ O)
{
  const int qt  = blockIdx.x;        // 0..7
  const int hh  = blockIdx.y;        // 0..31
  const int bo  = blockIdx.z;        // 0..3
  const int tid = threadIdx.x;
  const int wave = tid >> 5, lane = tid & 31;
  const int r  = lane & 15;
  const int hi = lane >> 4;

  __shared__ __align__(16) unsigned short sS[QT][NSEQ];   // scores -> probs (bf16)
  __shared__ __align__(16) unsigned short sVT[32][NSEQ];  // V transposed (d, key)
  __shared__ float red[QT][4];
  __shared__ float rowmax[QT];
  __shared__ float rowsum[QT];

  const int colbase = hh * 32;
  const int trow0   = bo * NSEQ;
  const int qrow0   = trow0 + qt * QT;

  // stage V^T
  for (int i = tid; i < NSEQ * 32; i += 256) {
    const int key = i >> 5, d = i & 31;
    sVT[d][key] = Vb[(size_t)(trow0 + key) * HD + colbase + d];
  }

  // ---- scores = Q K^T * 1/sqrt(dh), WMMA, written bf16 to sS ----
  const int mt = wave & 3;
  Frag16 a;
  {
    const unsigned short* qp = Q + (size_t)(qrow0 + mt * 16 + r) * HD + colbase;
    const int db = hi * 8;
    a.h[0] = *(const u16x8*)(qp + db);
    a.h[1] = *(const u16x8*)(qp + db + 16);
  }
  const float sscale = 0.17677669529663687f;  // 1/sqrt(32)
  for (int j = 0; j < 16; ++j) {
    const int nt = (wave >> 2) * 16 + j;
    const unsigned short* kp =
        Kb + (size_t)(trow0 + nt * 16 + r) * HD + colbase + hi * 8;
    Frag16 bf;
    bf.h[0] = *(const u16x8*)kp;
    bf.h[1] = *(const u16x8*)(kp + 16);
    v8f c = {};
    c = wmma_bf16(a, bf, c);
#pragma unroll
    for (int rr = 0; rr < 8; ++rr)
      sS[mt * 16 + rr + hi * 8][nt * 16 + r] = f2bf(c[rr] * sscale);
  }
  __syncthreads();

  // ---- softmax over 512 keys (4 segment-threads per row) ----
  const int srow = tid & 63;
  const int seg  = tid >> 6;
  const int cbeg = seg * 128;
  float mx = -3.0e38f;
  for (int c = 0; c < 128; ++c) mx = fmaxf(mx, bf2f(sS[srow][cbeg + c]));
  red[srow][seg] = mx; __syncthreads();
  if (seg == 0)
    rowmax[srow] = fmaxf(fmaxf(red[srow][0], red[srow][1]),
                         fmaxf(red[srow][2], red[srow][3]));
  __syncthreads();
  const float rm = rowmax[srow];
  float ssum = 0.f;
  for (int c = 0; c < 128; ++c) {
    const unsigned short eb = f2bf(__expf(bf2f(sS[srow][cbeg + c]) - rm));
    sS[srow][cbeg + c] = eb;
    ssum += bf2f(eb);
  }
  red[srow][seg] = ssum; __syncthreads();
  if (seg == 0)
    rowsum[srow] = red[srow][0] + red[srow][1] + red[srow][2] + red[srow][3];
  __syncthreads();

  // ---- O = P V, WMMA over 16 K-steps of 32 keys ----
  const int nt = wave >> 2;   // 0..1 (dh tiles)
  v8f oc = {};
  for (int kk = 0; kk < 16; ++kk) {
    Frag16 pa;
    const int kb = kk * 32 + hi * 8;
    pa.h[0] = *(const u16x8*)&sS[mt * 16 + r][kb];
    pa.h[1] = *(const u16x8*)&sS[mt * 16 + r][kb + 16];
    Frag16 vbf;
    const int kb2 = kk * 32 + hi * 8;
    vbf.h[0] = *(const u16x8*)&sVT[nt * 16 + r][kb2];
    vbf.h[1] = *(const u16x8*)&sVT[nt * 16 + r][kb2 + 16];
    oc = wmma_bf16(pa, vbf, oc);
  }
#pragma unroll
  for (int rr = 0; rr < 8; ++rr) {
    const int row = mt * 16 + rr + hi * 8;
    O[(size_t)(qrow0 + row) * HD + colbase + nt * 16 + r] =
        f2bf(oc[rr] / rowsum[row]);
  }
}

// ---------------- sparse gate: softmax(32), top-16, renorm, aux sums ----------
__global__ void gate_s_k(const float* __restrict__ ht, const float* __restrict__ gs,
                         float* __restrict__ probs, float* __restrict__ wcomb,
                         float* __restrict__ fsum, float* __restrict__ psum)
{
  const int t = blockIdx.x;
  const int e = threadIdx.x;   // 0..31
  const float* xp = ht + (size_t)t * HD;
  float acc = 0.f;
  for (int k = 0; k < HD; ++k) acc += xp[k] * gs[k * 32 + e];
  __shared__ float p[32];
  __shared__ float selv[32];
  p[e] = acc; __syncthreads();
  float mx = p[0];
  for (int j = 1; j < 32; ++j) mx = fmaxf(mx, p[j]);
  __syncthreads();
  const float ex = __expf(acc - mx);
  p[e] = ex; __syncthreads();
  float sum = 0.f;
  for (int j = 0; j < 32; ++j) sum += p[j];
  const float prob = ex / sum;
  probs[(size_t)t * 32 + e] = prob;
  // rank on ex (softmax monotonic); ties broken by lower index (lax.top_k)
  int rank = 0;
  for (int j = 0; j < 32; ++j)
    rank += (p[j] > ex) || (p[j] == ex && j < e);
  const bool sel = rank < 16;
  selv[e] = sel ? prob : 0.f; __syncthreads();
  float ss = 0.f;
  for (int j = 0; j < 32; ++j) ss += selv[j];
  wcomb[(size_t)t * 32 + e] = sel ? prob / ss : 0.f;
  atomicAdd(&psum[e], prob);
  if (sel) atomicAdd(&fsum[e], 1.0f);
}

// ---------------- dense gate: softmax over 4 ----------------
__global__ void gate_d_k(const float* __restrict__ ht, const float* __restrict__ gd,
                         float* __restrict__ pd)
{
  const int t = blockIdx.x;
  const int e = threadIdx.x;
  __shared__ float l[4];
  if (e < 4) {
    const float* xp = ht + (size_t)t * HD;
    float acc = 0.f;
    for (int k = 0; k < HD; ++k) acc += xp[k] * gd[k * 4 + e];
    l[e] = acc;
  }
  __syncthreads();
  if (e < 4) {
    const float mx = fmaxf(fmaxf(l[0], l[1]), fmaxf(l[2], l[3]));
    const float s = __expf(l[0] - mx) + __expf(l[1] - mx) +
                    __expf(l[2] - mx) + __expf(l[3] - mx);
    pd[(size_t)t * 4 + e] = __expf(l[e] - mx) / s;
  }
}

__global__ void aux_k(const float* __restrict__ fsum, const float* __restrict__ psum,
                      float* __restrict__ out)
{
  const int e = threadIdx.x;
  __shared__ float v[32];
  const float invT = 1.0f / (float)TOK;
  v[e] = (fsum[e] * invT) * (psum[e] * invT);
  __syncthreads();
  if (e == 0) {
    float s = 0.f;
    for (int j = 0; j < 32; ++j) s += v[j];
    *out = 32.0f * s;
  }
}

__global__ void zero_k(float* __restrict__ p, int n)
{
  const int i = blockIdx.x * blockDim.x + threadIdx.x;
  if (i < n) p[i] = 0.f;
}

// ---------------- host orchestration ----------------
extern "C" void kernel_launch(void* const* d_in, const int* in_sizes, int n_in,
                              void* d_out, int out_size, void* d_ws, size_t ws_size,
                              hipStream_t stream)
{
  (void)in_sizes; (void)n_in; (void)out_size; (void)ws_size;
  const float* x      = (const float*)d_in[0];
  const float* ln_g   = (const float*)d_in[1];
  const float* ln_b   = (const float*)d_in[2];
  const float* wq = (const float*)d_in[3];  const float* bq = (const float*)d_in[4];
  const float* wk = (const float*)d_in[5];  const float* bk = (const float*)d_in[6];
  const float* wv = (const float*)d_in[7];  const float* bv = (const float*)d_in[8];
  const float* wo = (const float*)d_in[9];  const float* bo = (const float*)d_in[10];
  const float* gate_s = (const float*)d_in[11];
  const float* w1_s = (const float*)d_in[12]; const float* b1_s = (const float*)d_in[13];
  const float* w2_s = (const float*)d_in[14]; const float* b2_s = (const float*)d_in[15];
  const float* gate_d = (const float*)d_in[16];
  const float* w1_d = (const float*)d_in[17]; const float* b1_d = (const float*)d_in[18];
  const float* w2_d = (const float*)d_in[19]; const float* b2_d = (const float*)d_in[20];

  float* out = (float*)d_out;

  char* wsb = (char*)d_ws;
  size_t off = 0;
  auto alloc = [&](size_t bytes) -> char* {
    char* p = wsb + off;
    off += (bytes + 255) & ~(size_t)255;
    return p;
  };
  typedef unsigned short u16;
  u16*   h1b   = (u16*)alloc((size_t)TOK * HD * 2);
  u16*   qb    = (u16*)alloc((size_t)TOK * HD * 2);
  u16*   kb    = (u16*)alloc((size_t)TOK * HD * 2);
  u16*   vb    = (u16*)alloc((size_t)TOK * HD * 2);
  u16*   ob    = (u16*)alloc((size_t)TOK * HD * 2);
  float* h2f   = (float*)alloc((size_t)TOK * HD * 4);
  u16*   h2b   = (u16*)alloc((size_t)TOK * HD * 2);
  u16*   A1b   = (u16*)alloc((size_t)TOK * EHD * 2);
  float* probs = (float*)alloc((size_t)TOK * 32 * 4);
  float* wcomb = (float*)alloc((size_t)TOK * 32 * 4);
  float* pdw   = (float*)alloc((size_t)TOK * 4 * 4);
  float* fsum  = (float*)alloc(64 * 4);
  float* psum  = fsum + 32;
  // pre-transposed bf16 weights
  u16* wqt  = (u16*)alloc((size_t)HD * HD * 2);
  u16* wkt  = (u16*)alloc((size_t)HD * HD * 2);
  u16* wvt  = (u16*)alloc((size_t)HD * HD * 2);
  u16* wot  = (u16*)alloc((size_t)HD * HD * 2);
  u16* w1st = (u16*)alloc((size_t)32 * HD * EHD * 2);
  u16* w2st = (u16*)alloc((size_t)32 * EHD * HD * 2);
  u16* w1dt = (u16*)alloc((size_t)4 * HD * EHD * 2);
  u16* w2dt = (u16*)alloc((size_t)4 * EHD * HD * 2);

  const dim3 blk(256);

  // ---- one-time (per launch) weight convert + transpose ----
  wT_k<<<dim3(HD / 32, HD / 32, 1), blk, 0, stream>>>(wq, wqt, HD, HD);
  wT_k<<<dim3(HD / 32, HD / 32, 1), blk, 0, stream>>>(wk, wkt, HD, HD);
  wT_k<<<dim3(HD / 32, HD / 32, 1), blk, 0, stream>>>(wv, wvt, HD, HD);
  wT_k<<<dim3(HD / 32, HD / 32, 1), blk, 0, stream>>>(wo, wot, HD, HD);
  wT_k<<<dim3(EHD / 32, HD / 32, 32), blk, 0, stream>>>(w1_s, w1st, HD, EHD);
  wT_k<<<dim3(HD / 32, EHD / 32, 32), blk, 0, stream>>>(w2_s, w2st, EHD, HD);
  wT_k<<<dim3(EHD / 32, HD / 32, 4), blk, 0, stream>>>(w1_d, w1dt, HD, EHD);
  wT_k<<<dim3(HD / 32, EHD / 32, 4), blk, 0, stream>>>(w2_d, w2dt, EHD, HD);

  const dim3 gHH(HD / BN, TOK / BM);

  // ---- attention block ----
  ln_k<<<TOK, blk, 0, stream>>>(x, ln_g, ln_b, nullptr, h1b);
  gemm_k<F_OUT_BF16><<<gHH, blk, 0, stream>>>(h1b, wqt, bq, nullptr, 0, nullptr,
                                              nullptr, qb, TOK, HD, HD);
  gemm_k<F_OUT_BF16><<<gHH, blk, 0, stream>>>(h1b, wkt, bk, nullptr, 0, nullptr,
                                              nullptr, kb, TOK, HD, HD);
  gemm_k<F_OUT_BF16><<<gHH, blk, 0, stream>>>(h1b, wvt, bv, nullptr, 0, nullptr,
                                              nullptr, vb, TOK, HD, HD);
  attn_k<<<dim3(NSEQ / QT, NHEADS, NOBJS), blk, 0, stream>>>(qb, kb, vb, ob);
  // x1 = o @ wo + bo + x  -> written straight into d_out (this is identity2)
  gemm_k<F_ADDSRC><<<gHH, blk, 0, stream>>>(ob, wot, bo, nullptr, 0, x,
                                            out, nullptr, TOK, HD, HD);

  // ---- MoE gating ----
  ln_k<<<TOK, blk, 0, stream>>>(out, ln_g, ln_b, h2f, h2b);
  zero_k<<<1, 64, 0, stream>>>(fsum, 64);
  gate_s_k<<<TOK, 32, 0, stream>>>(h2f, gate_s, probs, wcomb, fsum, psum);
  gate_d_k<<<TOK, 32, 0, stream>>>(h2f, gate_d, pdw);
  aux_k<<<1, 32, 0, stream>>>(fsum, psum, out + (size_t)TOK * HD);

  // ---- expert FFNs: out += (scale .* gelu(h2@W1+b1)) @ W2 + scale .* b2 ----
  const dim3 g1(EHD / BN, TOK / BM);
  const dim3 g2(HD / BN, TOK / BM);
  for (int e = 0; e < 32; ++e) {
    gemm_k<F_GELU | F_SCALE_AFTER | F_OUT_BF16><<<g1, blk, 0, stream>>>(
        h2b, w1st + (size_t)e * HD * EHD, b1_s + (size_t)e * EHD,
        wcomb + e, 32, nullptr, nullptr, A1b, TOK, EHD, HD);
    gemm_k<F_SCALED_BIAS | F_ADDSRC><<<g2, blk, 0, stream>>>(
        A1b, w2st + (size_t)e * EHD * HD, b2_s + (size_t)e * HD,
        wcomb + e, 32, out, out, nullptr, TOK, HD, EHD);
  }
  for (int e = 0; e < 4; ++e) {
    gemm_k<F_GELU | F_SCALE_AFTER | F_OUT_BF16><<<g1, blk, 0, stream>>>(
        h2b, w1dt + (size_t)e * HD * EHD, b1_d + (size_t)e * EHD,
        pdw + e, 4, nullptr, nullptr, A1b, TOK, EHD, HD);
    gemm_k<F_SCALED_BIAS | F_ADDSRC><<<g2, blk, 0, stream>>>(
        A1b, w2dt + (size_t)e * EHD * HD, b2_d + (size_t)e * HD,
        pdw + e, 4, out, out, nullptr, TOK, HD, EHD);
  }
}
